// H2GCN_90400471646625
// MI455X (gfx1250) — matrix operand order repaired and checked by reading, not compiled, not run on hardware
//
#include <hip/hip_runtime.h>

typedef __attribute__((ext_vector_type(16))) _Float16 v16h;
typedef __attribute__((ext_vector_type(4)))  _Float16 v4h;
typedef __attribute__((ext_vector_type(8)))  float    v8f;

// ---------------------------------------------------------------------------
// zero-fill (x1/x2 accumulators must start at 0 every call)
// ---------------------------------------------------------------------------
__global__ __launch_bounds__(256) void zero_kernel(float4* __restrict__ p, long long n4) {
    long long i = (long long)blockIdx.x * 256 + threadIdx.x;
    long long stride = (long long)gridDim.x * 256;
    for (; i < n4; i += stride) p[i] = make_float4(0.f, 0.f, 0.f, 0.f);
}

// Fragment-major index helper (wave32 WMMA 16-bit A layout, ISA 7.12.2)
// A 16x32 (MxK): lane = 16*((k>>3)&1) + m ; e = (k' < 8) ? k' : k'-8, k'=k-8g
__device__ __forceinline__ void a_frag_idx(int m, int k, int& lane, int& e) {
    const int g = (k >> 3) & 1;
    const int kp = k - 8 * g;
    lane = 16 * g + m;
    e = (kp < 8) ? kp : (kp - 8);
}

__device__ __forceinline__ v4h cvt4(const float4 v) {
    return (v4h){ (_Float16)v.x, (_Float16)v.y, (_Float16)v.z, (_Float16)v.w };
}

// ---------------------------------------------------------------------------
// GEMM1: x0 = relu(x[N,512] @ W1[512,64] + b1)   (WMMA f32<-f16, wave32)
// 256 threads = 8 waves; wave owns 16x64 tile; block = 128 rows.
// LDS fragment-major; register double-buffered staging pipeline.
// ---------------------------------------------------------------------------
__global__ __launch_bounds__(256) void gemm1_kernel(const float* __restrict__ x,
                                                    const float* __restrict__ W1,
                                                    const float* __restrict__ b1,
                                                    float* __restrict__ x0) {
    __shared__ __align__(16) _Float16 As[8][32][16];   // 8 KB
    __shared__ __align__(16) _Float16 Bs[4][32][16];   // 4 KB

    const int t    = threadIdx.x;
    const int lane = t & 31;
    const int w    = t >> 5;
    const int g    = lane >> 4;
    const int nn   = lane & 15;
    const int r0   = blockIdx.x * 128;

    // loop-invariant staging geometry
    int arow[4]; int acol[4]; v4h* adst[4];
#pragma unroll
    for (int it = 0; it < 4; ++it) {
        const int i = (it * 256 + t) * 4;
        arow[it] = i >> 5;  acol[it] = i & 31;
        int fl, fe; a_frag_idx(arow[it] & 15, acol[it], fl, fe);
        adst[it] = (v4h*)&As[arow[it] >> 4][fl][fe];
    }
    int bcol[2]; int bkr[2]; v4h* bdst[2];
#pragma unroll
    for (int it = 0; it < 2; ++it) {
        const int idx = it * 256 + t;
        bcol[it] = idx & 63;  bkr[it] = (idx >> 6) * 4;
        bdst[it] = (v4h*)&Bs[bcol[it] >> 4][16 * ((bkr[it] >> 3) & 1) + (bcol[it] & 15)][(bkr[it] & 7) + 8 * (bkr[it] >> 4)];
    }

    float4 aReg[4]; float4 bReg[2];

    auto load_tile = [&](int ks) {
        const int kb = ks * 32;
#pragma unroll
        for (int it = 0; it < 4; ++it) {
            const float* p = &x[(size_t)(r0 + arow[it]) * 512 + kb + acol[it]];
            aReg[it] = *(const float4*)p;
            if (ks + 2 < 16) __builtin_prefetch(p + 64, 0, 3);   // warm K+2 slice
        }
#pragma unroll
        for (int it = 0; it < 2; ++it) {
            const float* p = &W1[(size_t)(kb + bkr[it]) * 64 + bcol[it]];
            bReg[it] = make_float4(p[0], p[64], p[128], p[192]);
        }
    };
    auto store_tile = [&]() {
#pragma unroll
        for (int it = 0; it < 4; ++it) *adst[it] = cvt4(aReg[it]);
#pragma unroll
        for (int it = 0; it < 2; ++it) *bdst[it] = cvt4(bReg[it]);
    };

    v8f acc0 = {}, acc1 = {}, acc2 = {}, acc3 = {};

    load_tile(0);
    for (int ks = 0; ks < 16; ++ks) {
        store_tile();                 // waits on outstanding loads here only
        __syncthreads();
        if (ks + 1 < 16) load_tile(ks + 1);   // overlap next loads with WMMA

        const v16h a   = *(const v16h*)As[w][lane];
        const v16h b0  = *(const v16h*)Bs[0][lane];
        const v16h b1v = *(const v16h*)Bs[1][lane];
        const v16h b2  = *(const v16h*)Bs[2][lane];
        const v16h b3  = *(const v16h*)Bs[3][lane];

        acc0 = __builtin_amdgcn_wmma_f32_16x16x32_f16(false, a, false, b0,  (short)0, acc0, false, false);
        acc1 = __builtin_amdgcn_wmma_f32_16x16x32_f16(false, a, false, b1v, (short)0, acc1, false, false);
        acc2 = __builtin_amdgcn_wmma_f32_16x16x32_f16(false, a, false, b2,  (short)0, acc2, false, false);
        acc3 = __builtin_amdgcn_wmma_f32_16x16x32_f16(false, a, false, b3,  (short)0, acc3, false, false);
        __syncthreads();
    }

    const float bb0 = b1[nn], bb1 = b1[16 + nn], bb2 = b1[32 + nn], bb3 = b1[48 + nn];
#pragma unroll
    for (int r = 0; r < 8; ++r) {
        const int m = 8 * g + r;                    // C/D layout: M = 8g + r
        float* row = x0 + (size_t)(r0 + w * 16 + m) * 64;
        row[nn]      = fmaxf(acc0[r] + bb0, 0.f);
        row[16 + nn] = fmaxf(acc1[r] + bb1, 0.f);
        row[32 + nn] = fmaxf(acc2[r] + bb2, 0.f);
        row[48 + nn] = fmaxf(acc3[r] + bb3, 0.f);
    }
}

// ---------------------------------------------------------------------------
// SpMM scatter: dst[row] += val * src[col]   (F/4 lanes per edge, float4 gather,
// 4x FP32 global atomic scatter). shift = log2(F/4).
// ---------------------------------------------------------------------------
__global__ __launch_bounds__(256) void spmm_kernel(const int* __restrict__ rows,
                                                   const int* __restrict__ cols,
                                                   const float* __restrict__ vals, int nE,
                                                   const float* __restrict__ src, int srcStride, int srcOff,
                                                   float* __restrict__ dst, int dstStride, int dstOff,
                                                   int shift) {
    const long long t = (long long)blockIdx.x * 256 + threadIdx.x;
    const int e = (int)(t >> shift);
    if (e >= nE) return;
    const int q = (int)(t & ((1 << shift) - 1));
    const int f = q << 2;

    const int r = rows[e];
    const int c = cols[e];
    const float v = vals[e];

    const float4 xv = *(const float4*)(src + (size_t)c * srcStride + srcOff + f);
    float* dp = dst + (size_t)r * dstStride + dstOff + f;
    atomicAdd(dp + 0, v * xv.x);
    atomicAdd(dp + 1, v * xv.y);
    atomicAdd(dp + 2, v * xv.z);
    atomicAdd(dp + 3, v * xv.w);
}

// ---------------------------------------------------------------------------
// GEMM2: logits = feat[N,448] @ Wf[448,40] + bf   (WMMA, N padded 40->48)
// feat is virtual: cols [0,64)=x0, [64,192)=x1, [192,448)=x2
// ---------------------------------------------------------------------------
__global__ __launch_bounds__(256) void gemm2_kernel(const float* __restrict__ x0,
                                                    const float* __restrict__ x1,
                                                    const float* __restrict__ x2,
                                                    const float* __restrict__ Wf,
                                                    const float* __restrict__ bf,
                                                    float* __restrict__ logits) {
    __shared__ __align__(16) _Float16 As[8][32][16];   // 8 KB
    __shared__ __align__(16) _Float16 Bs[3][32][16];   // 3 KB

    const int t    = threadIdx.x;
    const int lane = t & 31;
    const int w    = t >> 5;
    const int g    = lane >> 4;
    const int nn   = lane & 15;
    const int r0   = blockIdx.x * 128;

    int arow[4]; int acol[4]; v4h* adst[4];
#pragma unroll
    for (int it = 0; it < 4; ++it) {
        const int i = (it * 256 + t) * 4;
        arow[it] = i >> 5;  acol[it] = i & 31;
        int fl, fe; a_frag_idx(arow[it] & 15, acol[it], fl, fe);
        adst[it] = (v4h*)&As[arow[it] >> 4][fl][fe];
    }
    const bool bact = (t < 384);
    const int bcol = t % 48;
    const int bkr  = (t / 48) * 4;
    v4h* bdst = (v4h*)&Bs[(bcol >> 4) & 3][16 * ((bkr >> 3) & 1) + (bcol & 15)][(bkr & 7) + 8 * (bkr >> 4)];

    float4 aReg[4]; float4 bReg;

    auto load_tile = [&](int ks) {
        const int kb = ks * 32;
        const float* src; int stride, coff;
        if (kb < 64)       { src = x0; stride = 64;  coff = kb; }
        else if (kb < 192) { src = x1; stride = 128; coff = kb - 64; }
        else               { src = x2; stride = 256; coff = kb - 192; }
#pragma unroll
        for (int it = 0; it < 4; ++it)
            aReg[it] = *(const float4*)&src[(size_t)(r0 + arow[it]) * stride + coff + acol[it]];
        if (bact && bcol < 40) {
            const float* p = &Wf[(size_t)(kb + bkr) * 40 + bcol];
            bReg = make_float4(p[0], p[40], p[80], p[120]);
        } else {
            bReg = make_float4(0.f, 0.f, 0.f, 0.f);
        }
    };
    auto store_tile = [&]() {
#pragma unroll
        for (int it = 0; it < 4; ++it) *adst[it] = cvt4(aReg[it]);
        if (bact) *bdst = cvt4(bReg);
    };

    v8f acc0 = {}, acc1 = {}, acc2 = {};

    load_tile(0);
    for (int ks = 0; ks < 14; ++ks) {
        store_tile();
        __syncthreads();
        if (ks + 1 < 14) load_tile(ks + 1);

        const v16h a   = *(const v16h*)As[w][lane];
        const v16h b0  = *(const v16h*)Bs[0][lane];
        const v16h b1v = *(const v16h*)Bs[1][lane];
        const v16h b2  = *(const v16h*)Bs[2][lane];

        acc0 = __builtin_amdgcn_wmma_f32_16x16x32_f16(false, a, false, b0,  (short)0, acc0, false, false);
        acc1 = __builtin_amdgcn_wmma_f32_16x16x32_f16(false, a, false, b1v, (short)0, acc1, false, false);
        acc2 = __builtin_amdgcn_wmma_f32_16x16x32_f16(false, a, false, b2,  (short)0, acc2, false, false);
        __syncthreads();
    }

    const float bb0 = bf[nn];
    const float bb1 = bf[16 + nn];
    const float bb2 = (nn < 8) ? bf[32 + nn] : 0.f;
#pragma unroll
    for (int r = 0; r < 8; ++r) {
        const int m = 8 * g + r;
        float* row = logits + (size_t)(r0 + w * 16 + m) * 40;
        row[nn]      = acc0[r] + bb0;
        row[16 + nn] = acc1[r] + bb1;
        if (nn < 8) row[32 + nn] = acc2[r] + bb2;
    }
}

// ---------------------------------------------------------------------------
// log_softmax over 40 logits: one wave32 per row, shfl_xor reductions
// ---------------------------------------------------------------------------
__global__ __launch_bounds__(256) void logsoftmax_kernel(const float* __restrict__ logits,
                                                         float* __restrict__ out) {
    const int row = blockIdx.x * 8 + (threadIdx.x >> 5);
    const int l   = threadIdx.x & 31;
    const float* p = logits + (size_t)row * 40;

    const float v0 = p[l];
    const float v1 = (l < 8) ? p[32 + l] : -3.402823466e+38f;

    float m = fmaxf(v0, v1);
#pragma unroll
    for (int s = 16; s > 0; s >>= 1) m = fmaxf(m, __shfl_xor(m, s));

    float sum = __expf(v0 - m) + ((l < 8) ? __expf(v1 - m) : 0.f);
#pragma unroll
    for (int s = 16; s > 0; s >>= 1) sum += __shfl_xor(sum, s);

    const float lse = m + __logf(sum);
    float* o = out + (size_t)row * 40;
    o[l] = v0 - lse;
    if (l < 8) o[32 + l] = v1 - lse;
}

// ---------------------------------------------------------------------------
// launch
// ---------------------------------------------------------------------------
extern "C" void kernel_launch(void* const* d_in, const int* in_sizes, int n_in,
                              void* d_out, int out_size, void* d_ws, size_t ws_size,
                              hipStream_t stream) {
    const float* x     = (const float*)d_in[0];
    const int*   rows0 = (const int*)  d_in[1];
    const int*   cols0 = (const int*)  d_in[2];
    const float* vals0 = (const float*)d_in[3];
    const int*   rows1 = (const int*)  d_in[4];
    const int*   cols1 = (const int*)  d_in[5];
    const float* vals1 = (const float*)d_in[6];
    const float* W1    = (const float*)d_in[7];
    const float* b1    = (const float*)d_in[8];
    const float* Wf    = (const float*)d_in[9];
    const float* bf    = (const float*)d_in[10];

    const int N = in_sizes[0] / 512;   // 65536
    const int E = in_sizes[1];         // 1048576

    float* ws     = (float*)d_ws;
    float* x0     = ws;                          // [N,64]
    float* x1     = ws + (size_t)N * 64;         // [N,128] = [S0@x0 | S1@x0]
    float* x2     = ws + (size_t)N * 192;        // [N,256] = [S0@x1 | S1@x1]
    float* logits = ws + (size_t)N * 448;        // [N,40]
    float* out    = (float*)d_out;

    // 1) zero the scatter accumulators (x1,x2 contiguous: N*384 floats)
    {
        long long n4 = (long long)N * 384 / 4;
        zero_kernel<<<4096, 256, 0, stream>>>((float4*)x1, n4);
    }

    // 2) x0 = relu(x @ W1 + b1)
    gemm1_kernel<<<N / 128, 256, 0, stream>>>(x, W1, b1, x0);

    // 3) first hop: width-64 SpMMs into x1 halves
    {
        int blocks = (int)(((long long)E * 16 + 255) / 256);
        spmm_kernel<<<blocks, 256, 0, stream>>>(rows0, cols0, vals0, E, x0, 64, 0, x1, 128, 0,  4);
        spmm_kernel<<<blocks, 256, 0, stream>>>(rows1, cols1, vals1, E, x0, 64, 0, x1, 128, 64, 4);
    }

    // 4) second hop: width-128 SpMMs over full x1 into x2 halves
    {
        int blocks = (int)(((long long)E * 32 + 255) / 256);
        spmm_kernel<<<blocks, 256, 0, stream>>>(rows0, cols0, vals0, E, x1, 128, 0, x2, 256, 0,   5);
        spmm_kernel<<<blocks, 256, 0, stream>>>(rows1, cols1, vals1, E, x1, 128, 0, x2, 256, 128, 5);
    }

    // 5) logits = feat @ Wf + bf
    gemm2_kernel<<<N / 128, 256, 0, stream>>>(x0, x1, x2, Wf, bf, logits);

    // 6) log_softmax
    logsoftmax_kernel<<<N / 8, 256, 0, stream>>>(logits, out);
}
